// GATPredictor_70643622084831
// MI455X (gfx1250) — matrix operand order repaired and checked by reading, not compiled, not run on hardware
//
#include <hip/hip_runtime.h>
#include <hip/hip_bf16.h>

typedef __attribute__((ext_vector_type(16))) _Float16 v16h;
typedef __attribute__((ext_vector_type(8)))  float    v8f;

#define FDIM      128   // F_IN == HID*HEADS == 128 for both layers
#define NHEADS    4
#define CHID      32
#define NGRAPHS   64
#define NEG_SLOPE 0.2f

// ---------------------------------------------------------------------------
// Dense projection: OUT[N,128] = IN[N,128] @ W[128,128], fp32 in/out,
// f16 WMMA inner product (f32 accumulate). 8 waves/block, each wave owns a
// 16-wide column strip; persistent blocks loop over 16-row tiles.
// ---------------------------------------------------------------------------
__global__ __launch_bounds__(256) void gemm_wmma(
    const float* __restrict__ X, const float* __restrict__ W,
    float* __restrict__ OUT, int nRows, int nTiles)
{
    const int wave   = threadIdx.x >> 5;
    const int lane   = threadIdx.x & 31;
    const int half16 = lane >> 4;      // 0: lanes 0-15, 1: lanes 16-31
    const int lmod   = lane & 15;
    const int n0     = wave * 16;      // column strip for this wave

    // Preload B fragments for the whole K=128 (4 chunks of 32).
    // 16-bit B 32x16 layout: VGPR v, pack p -> K = 2*(v%4) + 16*(v/4) + 8*half16 + p, N = lane%16
    v16h bfrag[4];
#pragma unroll
    for (int kc = 0; kc < 4; ++kc) {
#pragma unroll
        for (int v = 0; v < 8; ++v) {
#pragma unroll
            for (int p = 0; p < 2; ++p) {
                int k = kc * 32 + (v & 3) * 2 + (v >> 2) * 16 + half16 * 8 + p;
                bfrag[kc][2 * v + p] = (_Float16)W[k * FDIM + n0 + lmod];
            }
        }
    }

    for (int tile = blockIdx.x; tile < nTiles; tile += gridDim.x) {
        const int m0 = tile * 16;
        int m = m0 + lmod; if (m >= nRows) m = nRows - 1;   // clamp; store guarded
        const float* xr = X + (size_t)m * FDIM;

        v8f acc = {};
#pragma unroll
        for (int kc = 0; kc < 4; ++kc) {
            // 16-bit A 16x32 layout: VGPR v, pack p ->
            //   K = 2*(v%4) + 16*(v/4) + 8*half16 + p, M = lane%16
            v16h a;
#pragma unroll
            for (int v = 0; v < 8; ++v) {
#pragma unroll
                for (int p = 0; p < 2; ++p) {
                    int k = kc * 32 + (v & 3) * 2 + (v >> 2) * 16 + half16 * 8 + p;
                    a[2 * v + p] = (_Float16)xr[k];
                }
            }
            acc = __builtin_amdgcn_wmma_f32_16x16x32_f16(
                false, a, false, bfrag[kc], (short)0, acc, false, false);
        }

        // D f32 16x16: VGPR r -> row m0 + r + 8*half16, col n0 + lane%16
#pragma unroll
        for (int r = 0; r < 8; ++r) {
            int row = m0 + r + 8 * half16;
            if (row < nRows) OUT[(size_t)row * FDIM + n0 + lmod] = acc[r];
        }
    }
}

// ---------------------------------------------------------------------------
// Per-(node, head) attention logits: asrc/adst[n,h] = h[n,h,:] . att[h,:]
// ---------------------------------------------------------------------------
__global__ __launch_bounds__(256) void node_alpha(
    const float* __restrict__ hbuf,
    const float* __restrict__ att_src, const float* __restrict__ att_dst,
    float* __restrict__ asrc, float* __restrict__ adst, int nNodes)
{
    int idx = blockIdx.x * blockDim.x + threadIdx.x;
    if (idx >= nNodes * NHEADS) return;
    int n = idx >> 2, h = idx & 3;
    const float4* hp = (const float4*)(hbuf + (size_t)n * FDIM + h * CHID);
    const float4* as = (const float4*)(att_src + h * CHID);
    const float4* ad = (const float4*)(att_dst + h * CHID);
    float ts = 0.f, td = 0.f;
#pragma unroll
    for (int i = 0; i < CHID / 4; ++i) {
        float4 v = hp[i], s4 = as[i], d4 = ad[i];
        ts += v.x * s4.x + v.y * s4.y + v.z * s4.z + v.w * s4.w;
        td += v.x * d4.x + v.y * d4.y + v.z * d4.z + v.w * d4.w;
    }
    asrc[idx] = ts;
    adst[idx] = td;
}

// Order-preserving float -> u32 key for atomic segment-max.
__device__ __forceinline__ unsigned ordkey(float f) {
    unsigned u = __float_as_uint(f);
    return (u & 0x80000000u) ? ~u : (u | 0x80000000u);
}
__device__ __forceinline__ float unord(unsigned k) {
    return (k & 0x80000000u) ? __uint_as_float(k ^ 0x80000000u)
                             : __uint_as_float(~k);
}

__device__ __forceinline__ void edge_ends(const int* ei, int nEdges, int e,
                                          int& s, int& d) {
    if (e < nEdges) { s = ei[e]; d = ei[nEdges + e]; }
    else            { s = e - nEdges; d = s; }      // self-loop
}

// Pass 1: segment max of leaky_relu(asrc[src]+adst[dst]) over dst
__global__ __launch_bounds__(256) void edge_logit_max(
    const int* __restrict__ ei, int nEdges, int nNodes,
    const float* __restrict__ asrc, const float* __restrict__ adst,
    unsigned* __restrict__ mkey)
{
    int et = nEdges + nNodes;
    int idx = blockIdx.x * blockDim.x + threadIdx.x;
    if (idx >= et * NHEADS) return;
    int e = idx >> 2, h = idx & 3, s, d;
    edge_ends(ei, nEdges, e, s, d);
    float v = asrc[s * NHEADS + h] + adst[d * NHEADS + h];
    v = (v > 0.f) ? v : NEG_SLOPE * v;
    atomicMax(&mkey[d * NHEADS + h], ordkey(v));
}

// Pass 2: e = exp(v - max); store per-edge, segment-sum over dst
__global__ __launch_bounds__(256) void edge_exp_sum(
    const int* __restrict__ ei, int nEdges, int nNodes,
    const float* __restrict__ asrc, const float* __restrict__ adst,
    const unsigned* __restrict__ mkey,
    float* __restrict__ eexp, float* __restrict__ ssum)
{
    int et = nEdges + nNodes;
    int idx = blockIdx.x * blockDim.x + threadIdx.x;
    if (idx >= et * NHEADS) return;
    int e = idx >> 2, h = idx & 3, s, d;
    edge_ends(ei, nEdges, e, s, d);
    float v = asrc[s * NHEADS + h] + adst[d * NHEADS + h];
    v = (v > 0.f) ? v : NEG_SLOPE * v;
    float ex = __expf(v - unord(mkey[d * NHEADS + h]));
    eexp[idx] = ex;
    atomicAdd(&ssum[d * NHEADS + h], ex);
}

// Pass 3: out[dst, c] += h[src, c] * alpha.  2 edges per 256-thread block,
// one channel per thread; h and out both live in L2 (25.6 MB each).
__global__ __launch_bounds__(256) void gat_aggregate(
    const int* __restrict__ ei, int nEdges, int nNodes,
    const float* __restrict__ hsrc, const float* __restrict__ eexp,
    const float* __restrict__ ssum, float* __restrict__ outbuf)
{
    int et = nEdges + nNodes;
    int e = blockIdx.x * 2 + (threadIdx.x >> 7);
    if (e >= et) return;
    int c = threadIdx.x & 127, s, d;
    edge_ends(ei, nEdges, e, s, d);
    int h = c >> 5;
    float alpha = eexp[e * NHEADS + h] / (ssum[d * NHEADS + h] + 1e-16f);
    atomicAdd(&outbuf[(size_t)d * FDIM + c], hsrc[(size_t)s * FDIM + c] * alpha);
}

// Layer-1 epilogue: out = relu(in + b)
__global__ __launch_bounds__(256) void bias_relu(
    const float* __restrict__ in, const float* __restrict__ bias,
    float* __restrict__ out, int total)
{
    int i = blockIdx.x * blockDim.x + threadIdx.x;
    if (i >= total) return;
    out[i] = fmaxf(in[i] + bias[i & (FDIM - 1)], 0.f);
}

// mean-pool + fc folded: per node t = (h2[n]+b2).fc_w; reduce per graph.
__global__ __launch_bounds__(256) void node_pool(
    const float* __restrict__ h2, const float* __restrict__ b2,
    const float* __restrict__ fcw, const int* __restrict__ batch,
    float* __restrict__ gsum, float* __restrict__ gcnt, int nNodes)
{
    int n = blockIdx.x * blockDim.x + threadIdx.x;
    if (n >= nNodes) return;
    const float4* hp = (const float4*)(h2 + (size_t)n * FDIM);
    float t = 0.f;
#pragma unroll
    for (int i = 0; i < FDIM / 4; ++i) {
        float4 v = hp[i];
        float4 b = ((const float4*)b2)[i];
        float4 w = ((const float4*)fcw)[i];
        t += (v.x + b.x) * w.x + (v.y + b.y) * w.y +
             (v.z + b.z) * w.z + (v.w + b.w) * w.w;
    }
    int g = batch[n];
    atomicAdd(&gsum[g], t);
    atomicAdd(&gcnt[g], 1.0f);
}

__global__ void finalize(const float* __restrict__ gsum,
                         const float* __restrict__ gcnt,
                         const float* __restrict__ fcb, float* __restrict__ out)
{
    int g = threadIdx.x;
    if (g < NGRAPHS) out[g] = gsum[g] / fmaxf(gcnt[g], 1.0f) + fcb[0];
}

// ---------------------------------------------------------------------------
extern "C" void kernel_launch(void* const* d_in, const int* in_sizes, int n_in,
                              void* d_out, int out_size, void* d_ws, size_t ws_size,
                              hipStream_t stream)
{
    const float* x    = (const float*)d_in[0];
    const int*   ei   = (const int*)d_in[1];
    const int*   batch= (const int*)d_in[2];
    const float* W1   = (const float*)d_in[3];
    const float* as1  = (const float*)d_in[4];
    const float* ad1  = (const float*)d_in[5];
    const float* b1   = (const float*)d_in[6];
    const float* W2   = (const float*)d_in[7];
    const float* as2  = (const float*)d_in[8];
    const float* ad2  = (const float*)d_in[9];
    const float* b2   = (const float*)d_in[10];
    const float* fcw  = (const float*)d_in[11];
    const float* fcb  = (const float*)d_in[12];
    float* out = (float*)d_out;

    const int N  = in_sizes[0] / FDIM;   // 50000
    const int E  = in_sizes[1] / 2;      // 800000
    const int ET = E + N;                // + self-loops

    // --- workspace carve (all offsets 16B aligned) ---
    float*    bufA = (float*)d_ws;                        // N*128  (h / agg2)
    float*    bufB = bufA + (size_t)N * FDIM;             // N*128  (agg1 / h2)
    float*    eexp = bufB + (size_t)N * FDIM;             // ET*4
    float*    asrc = eexp + (size_t)ET * NHEADS;          // N*4
    float*    adst = asrc + (size_t)N * NHEADS;           // N*4
    unsigned* mkey = (unsigned*)(adst + (size_t)N * NHEADS); // N*4
    float*    ssum = (float*)mkey + (size_t)N * NHEADS;   // N*4
    float*    gsum = ssum + (size_t)N * NHEADS;           // 64
    float*    gcnt = gsum + NGRAPHS;                      // 64

    const int nTiles     = (N + 15) / 16;
    const int gemmBlocks = nTiles < 1024 ? nTiles : 1024;
    const int alphaBlks  = (N * NHEADS + 255) / 256;
    const int edgeBlks   = (ET * NHEADS + 255) / 256;
    const int aggBlks    = (ET + 1) / 2;
    const size_t nhB     = (size_t)N * NHEADS * sizeof(float);
    const size_t nodeB   = (size_t)N * FDIM * sizeof(float);

    // ================= layer 1 =================
    hipMemsetAsync(mkey, 0, nhB, stream);
    hipMemsetAsync(ssum, 0, nhB, stream);
    gemm_wmma<<<gemmBlocks, 256, 0, stream>>>(x, W1, bufA, N, nTiles);
    node_alpha<<<alphaBlks, 256, 0, stream>>>(bufA, as1, ad1, asrc, adst, N);
    edge_logit_max<<<edgeBlks, 256, 0, stream>>>(ei, E, N, asrc, adst, mkey);
    edge_exp_sum<<<edgeBlks, 256, 0, stream>>>(ei, E, N, asrc, adst, mkey, eexp, ssum);
    hipMemsetAsync(bufB, 0, nodeB, stream);
    gat_aggregate<<<aggBlks, 256, 0, stream>>>(ei, E, N, bufA, eexp, ssum, bufB);
    bias_relu<<<(N * FDIM + 255) / 256, 256, 0, stream>>>(bufB, b1, bufA, N * FDIM);

    // ================= layer 2 =================
    hipMemsetAsync(mkey, 0, nhB, stream);
    hipMemsetAsync(ssum, 0, nhB, stream);
    gemm_wmma<<<gemmBlocks, 256, 0, stream>>>(bufA, W2, bufB, N, nTiles);
    node_alpha<<<alphaBlks, 256, 0, stream>>>(bufB, as2, ad2, asrc, adst, N);
    edge_logit_max<<<edgeBlks, 256, 0, stream>>>(ei, E, N, asrc, adst, mkey);
    edge_exp_sum<<<edgeBlks, 256, 0, stream>>>(ei, E, N, asrc, adst, mkey, eexp, ssum);
    hipMemsetAsync(bufA, 0, nodeB, stream);
    gat_aggregate<<<aggBlks, 256, 0, stream>>>(ei, E, N, bufB, eexp, ssum, bufA);

    // ============ pool + fc ============
    hipMemsetAsync(gsum, 0, NGRAPHS * sizeof(float), stream);
    hipMemsetAsync(gcnt, 0, NGRAPHS * sizeof(float), stream);
    node_pool<<<(N + 255) / 256, 256, 0, stream>>>(bufA, b2, fcw, batch, gsum, gcnt, N);
    finalize<<<1, 64, 0, stream>>>(gsum, gcnt, fcb, out);
}
